// SelfAttention_43765716747093
// MI455X (gfx1250) — matrix operand order repaired and checked by reading, not compiled, not run on hardware
//
#include <hip/hip_runtime.h>
#include <hip/hip_bf16.h>

// ---------------------------------------------------------------------------
// Self-attention (B=4, N=2048, D=1024), bf16 WMMA pipeline for gfx1250.
//   1) cvt:   f32 -> bf16 for x, Wq, Wk, Wv
//   2) proj:  Q = x Wq^T, K = x Wk^T (row-major bf16), V^T = (x Wv^T)^T
//   3) score: P = softmax(Q K^T) / 32   (fused row softmax, bf16 out)
//   4) av:    out = P V                 (f32 out)
// GEMMs: v_wmma_f32_16x16x32_bf16 with LDS-staged, double-buffered tiles
// fed by CDNA5 async global->LDS copies (ASYNCcnt) when available.
// ---------------------------------------------------------------------------

#define BB 4
#define NN 2048
#define DD 1024
#define LDST 40          // LDS tile row stride (elements): 80B, bank-conflict free

typedef __attribute__((ext_vector_type(16))) __bf16        v16bf;
typedef __attribute__((ext_vector_type(8)))  float         v8f;
typedef __attribute__((ext_vector_type(4)))  unsigned int  u32x4;
typedef __attribute__((ext_vector_type(4)))  int           v4i;

union FragU {
  v16bf v;
  u32x4 q[2];
  unsigned short s[16];
};

#if defined(__has_builtin)
#if __has_builtin(__builtin_amdgcn_global_load_async_to_lds_b128) && \
    __has_builtin(__builtin_amdgcn_s_wait_asynccnt)
#define USE_ASYNC_LDS 1
#endif
#endif
#ifndef USE_ASYNC_LDS
#define USE_ASYNC_LDS 0
#endif

#if USE_ASYNC_LDS
typedef __attribute__((address_space(1))) v4i* as1p;   // global int4*
typedef __attribute__((address_space(3))) v4i* as3p;   // LDS int4*
#endif

__device__ __forceinline__ unsigned short f2bf(float f) {
  union { float f; unsigned int u; } c; c.f = f;
  unsigned int u = c.u;
  u += 0x7FFFu + ((u >> 16) & 1u);      // round-to-nearest-even
  return (unsigned short)(u >> 16);
}

// Copy 16 bf16 (32B) global -> LDS, async when the toolchain has the builtin.
__device__ __forceinline__ void async_copy16(const unsigned short* gp,
                                             unsigned short* lp) {
#if USE_ASYNC_LDS
  __builtin_amdgcn_global_load_async_to_lds_b128((as1p)gp,       (as3p)lp,       0, 0);
  __builtin_amdgcn_global_load_async_to_lds_b128((as1p)(gp + 8), (as3p)(lp + 8), 0, 0);
#else
  *reinterpret_cast<u32x4*>(lp)     = *reinterpret_cast<const u32x4*>(gp);
  *reinterpret_cast<u32x4*>(lp + 8) = *reinterpret_cast<const u32x4*>(gp + 8);
#endif
}

__device__ __forceinline__ void wait_async_all() {
#if USE_ASYNC_LDS
  __builtin_amdgcn_s_wait_asynccnt(0);
#endif
}

// A fragment: 16x32 bf16 tile, rows row0..row0+15, K = k0..k0+31.
// Lane L<16: row L, K k0+0..7 (VGPR0-3), k0+16..23 (VGPR4-7).
// Lane L>=16: row L-16, K k0+8..15 and k0+24..31.
__device__ __forceinline__ v16bf load_a_frag(const unsigned short* base, int ld,
                                             int row0, int k0, int lane) {
  int r  = row0 + (lane & 15);
  int kk = k0 + ((lane >> 4) << 3);
  const unsigned short* p = base + (size_t)r * ld + kk;
  FragU f;
  f.q[0] = *reinterpret_cast<const u32x4*>(p);
  f.q[1] = *reinterpret_cast<const u32x4*>(p + 16);
  return f.v;
}

// B fragment: 32x16 tile B[k][n], source row-per-n contiguous over k.
// Lane L<16: col L, K k0+0..7 and k0+16..23; lane L>=16: col L-16, K k0+8..15 and k0+24..31.
__device__ __forceinline__ v16bf load_b_frag(const unsigned short* base, int ld,
                                             int col0, int k0, int lane) {
  int c  = col0 + (lane & 15);
  int kk = k0 + ((lane >> 4) << 3);
  const unsigned short* p = base + (size_t)c * ld + kk;
  FragU f;
  f.q[0] = *reinterpret_cast<const u32x4*>(p);
  f.q[1] = *reinterpret_cast<const u32x4*>(p + 16);
  return f.v;
}

__device__ __forceinline__ v8f wmma_bf16(v16bf a, v16bf b, v8f c) {
  return __builtin_amdgcn_wmma_f32_16x16x32_bf16(false, a, false, b,
                                                 (short)0, c, false, false);
}

// Stage a 128x32 bf16 tile (rows row0.., cols kk..kk+31) into LDS (stride LDST).
// 256 threads, 2 per row, 16 elements each.
__device__ __forceinline__ void stage_tile(const unsigned short* g, int ld,
                                           int row0, int kk,
                                           unsigned short* lds, int t) {
  int r = t >> 1;
  int c = (t & 1) << 4;
  const unsigned short* gp = g + (size_t)(row0 + r) * ld + kk + c;
  async_copy16(gp, lds + r * LDST + c);
}

// Shared 128x128 GEMM core: C[2][4] per wave, double-buffered LDS tiles.
// A: [*, Kdim] rows Ablk..Ablk+127 ; B: rows (cols of C) Bblk..Bblk+127 over Kdim.
__device__ __forceinline__ void gemm128(const unsigned short* __restrict__ A,
                                        const unsigned short* __restrict__ B,
                                        int Kdim, int Ablk, int Bblk,
                                        unsigned short* ldsA, unsigned short* ldsB,
                                        v8f acc[2][4]) {
  const int t    = threadIdx.x;
  const int lane = t & 31;
  const int w    = t >> 5;
  const int m0   = (w & 3) * 32;      // wave offsets inside the 128x128 tile
  const int e0   = (w >> 2) * 64;
  const int BUF  = 128 * LDST;

  stage_tile(A, Kdim, Ablk, 0, ldsA, t);
  stage_tile(B, Kdim, Bblk, 0, ldsB, t);

  const int nstep = Kdim / 32;
  for (int s = 0; s < nstep; ++s) {
    const int cur = (s & 1) * BUF;
    const int nxt = ((s + 1) & 1) * BUF;
    wait_async_all();
    __syncthreads();                        // staged tiles visible to all waves
    if (s + 1 < nstep) {                    // overlap next copies with WMMAs
      stage_tile(A, Kdim, Ablk, (s + 1) * 32, ldsA + nxt, t);
      stage_tile(B, Kdim, Bblk, (s + 1) * 32, ldsB + nxt, t);
    }
    const unsigned short* Xs = ldsA + cur;
    const unsigned short* Ws = ldsB + cur;
    v16bf a0 = load_a_frag(Xs, LDST, m0,      0, lane);
    v16bf a1 = load_a_frag(Xs, LDST, m0 + 16, 0, lane);
    #pragma unroll
    for (int j = 0; j < 4; ++j) {
      v16bf bf = load_b_frag(Ws, LDST, e0 + j * 16, 0, lane);
      acc[0][j] = wmma_bf16(a0, bf, acc[0][j]);
      acc[1][j] = wmma_bf16(a1, bf, acc[1][j]);
    }
    __syncthreads();                        // reads done before buffer reuse
  }
}

// ---------------------------------------------------------------------------
// Kernel 1: f32 -> bf16 convert, 4 elements/thread
// ---------------------------------------------------------------------------
__global__ void cvt_f32_bf16(const float* __restrict__ src,
                             unsigned short* __restrict__ dst, int n4) {
  int i = blockIdx.x * blockDim.x + threadIdx.x;
  if (i < n4) {
    float4 v = reinterpret_cast<const float4*>(src)[i];
    unsigned int w0 = (unsigned)f2bf(v.x) | ((unsigned)f2bf(v.y) << 16);
    unsigned int w1 = (unsigned)f2bf(v.z) | ((unsigned)f2bf(v.w) << 16);
    reinterpret_cast<uint2*>(dst)[i] = make_uint2(w0, w1);
  }
}

// ---------------------------------------------------------------------------
// Kernel 2: projections. 128x128 block tile, 8 waves (4x2), 32x64 per wave.
// ---------------------------------------------------------------------------
__global__ __launch_bounds__(256) void proj_kernel(
    const unsigned short* __restrict__ xb,
    const unsigned short* __restrict__ wqb,
    const unsigned short* __restrict__ wkb,
    const unsigned short* __restrict__ wvb,
    unsigned short* __restrict__ Qb,
    unsigned short* __restrict__ Kb,
    unsigned short* __restrict__ Vtb) {
  __shared__ unsigned short ldsA[2 * 128 * LDST];
  __shared__ unsigned short ldsB[2 * 128 * LDST];

  const int lane = threadIdx.x & 31;
  const int w    = threadIdx.x >> 5;
  const int z    = blockIdx.z;
  const int b    = z / 3;
  const int mat  = z % 3;                       // 0=Q 1=K 2=V
  const unsigned short* W = (mat == 0) ? wqb : (mat == 1) ? wkb : wvb;
  const unsigned short* X = xb + (size_t)b * NN * DD;

  v8f zero = {};
  v8f acc[2][4];
  #pragma unroll
  for (int i = 0; i < 2; ++i)
    #pragma unroll
    for (int j = 0; j < 4; ++j) acc[i][j] = zero;

  gemm128(X, W, DD, blockIdx.x * 128, blockIdx.y * 128, ldsA, ldsB, acc);

  const int lh = lane >> 4;
  const int ln = lane & 15;
  const int m0 = blockIdx.x * 128 + (w & 3) * 32;
  const int e0 = blockIdx.y * 128 + (w >> 2) * 64;
  if (mat < 2) {
    unsigned short* O = ((mat == 0) ? Qb : Kb) + (size_t)b * NN * DD;
    #pragma unroll
    for (int i = 0; i < 2; ++i)
      #pragma unroll
      for (int j = 0; j < 4; ++j) {
        int e = e0 + j * 16 + ln;
        #pragma unroll
        for (int v = 0; v < 8; ++v) {
          int m = m0 + i * 16 + lh * 8 + v;
          O[(size_t)m * DD + e] = f2bf(acc[i][j][v]);
        }
      }
  } else {
    // V stored transposed: Vt[e][n] -> packed b128 stores along n.
    unsigned short* O = Vtb + (size_t)b * DD * NN;
    #pragma unroll
    for (int i = 0; i < 2; ++i)
      #pragma unroll
      for (int j = 0; j < 4; ++j) {
        int e     = e0 + j * 16 + ln;
        int mbase = m0 + i * 16 + lh * 8;
        u32x4 pk;
        #pragma unroll
        for (int v = 0; v < 4; ++v)
          pk[v] = (unsigned)f2bf(acc[i][j][2 * v]) |
                  ((unsigned)f2bf(acc[i][j][2 * v + 1]) << 16);
        *reinterpret_cast<u32x4*>(O + (size_t)e * NN + mbase) = pk;
      }
  }
}

// ---------------------------------------------------------------------------
// Kernel 3: fused S = Q K^T, row softmax, P = softmax(S)/32 as bf16.
// Workgroup = batch x 16 query rows x all 2048 keys; wave w keys [256w,256w+256).
// Q strip (16x1024) staged once in LDS (stride 1032 -> conflict free).
// ---------------------------------------------------------------------------
#define QSTRIDE 1032
__global__ __launch_bounds__(256) void score_softmax_kernel(
    const unsigned short* __restrict__ Qb,
    const unsigned short* __restrict__ Kb,
    unsigned short* __restrict__ Pb) {
  __shared__ unsigned short Qs[16 * QSTRIDE];
  __shared__ float redmax[8][16];
  __shared__ float redsum[8][16];

  const int t    = threadIdx.x;
  const int lane = t & 31;
  const int w    = t >> 5;
  const int b    = blockIdx.y;
  const int row0 = blockIdx.x * 16;
  const unsigned short* Q = Qb + (size_t)b * NN * DD;
  const unsigned short* K = Kb + (size_t)b * NN * DD;
  unsigned short*       P = Pb + (size_t)b * NN * NN;

  // Stage Q strip: 16 rows x 1024, 64 elements per thread.
  {
    int r = t >> 4;
    int c = (t & 15) * 64;
    const unsigned short* gq = Q + (size_t)(row0 + r) * DD + c;
    unsigned short*       lq = Qs + r * QSTRIDE + c;
    #pragma unroll
    for (int i = 0; i < 4; ++i) async_copy16(gq + i * 16, lq + i * 16);
  }
  wait_async_all();
  __syncthreads();

  const int n0 = w * 256;
  v8f zero = {};
  v8f acc[16];
  #pragma unroll
  for (int tt = 0; tt < 16; ++tt) acc[tt] = zero;

  for (int kk = 0; kk < DD; kk += 32) {
    v16bf qa = load_a_frag(Qs, QSTRIDE, 0, kk, lane);
    #pragma unroll
    for (int tt = 0; tt < 16; ++tt) {
      v16bf kb = load_b_frag(K, DD, n0 + tt * 16, kk, lane);
      acc[tt] = wmma_bf16(qa, kb, acc[tt]);
    }
  }

  const int lh = lane >> 4;
  const int ln = lane & 15;

  // Per-wave row max: row (lh*8+v) lives in VGPR v across one 16-lane half.
  float mv[8];
  #pragma unroll
  for (int v = 0; v < 8; ++v) {
    float m = acc[0][v];
    #pragma unroll
    for (int tt = 1; tt < 16; ++tt) m = fmaxf(m, acc[tt][v]);
    #pragma unroll
    for (int off = 1; off < 16; off <<= 1)
      m = fmaxf(m, __shfl_xor(m, off, 32));
    mv[v] = m;
  }
  if (ln == 0) {
    #pragma unroll
    for (int v = 0; v < 8; ++v) redmax[w][lh * 8 + v] = mv[v];
  }
  __syncthreads();
  float gmax[8];
  #pragma unroll
  for (int v = 0; v < 8; ++v) {
    float m = redmax[0][lh * 8 + v];
    #pragma unroll
    for (int ww = 1; ww < 8; ++ww) m = fmaxf(m, redmax[ww][lh * 8 + v]);
    gmax[v] = m;
  }

  // exp + per-wave row sums
  float sv[8];
  #pragma unroll
  for (int v = 0; v < 8; ++v) sv[v] = 0.0f;
  #pragma unroll
  for (int tt = 0; tt < 16; ++tt)
    #pragma unroll
    for (int v = 0; v < 8; ++v) {
      float e = __expf(acc[tt][v] - gmax[v]);
      acc[tt][v] = e;
      sv[v] += e;
    }
  #pragma unroll
  for (int v = 0; v < 8; ++v) {
    float s = sv[v];
    #pragma unroll
    for (int off = 1; off < 16; off <<= 1) s += __shfl_xor(s, off, 32);
    sv[v] = s;
  }
  if (ln == 0) {
    #pragma unroll
    for (int v = 0; v < 8; ++v) redsum[w][lh * 8 + v] = sv[v];
  }
  __syncthreads();
  #pragma unroll
  for (int v = 0; v < 8; ++v) {
    float s = 0.0f;
    #pragma unroll
    for (int ww = 0; ww < 8; ++ww) s += redsum[ww][lh * 8 + v];
    sv[v] = 1.0f / (s * 32.0f);   // fold post-softmax 1/sqrt(D) = 1/32
  }

  #pragma unroll
  for (int tt = 0; tt < 16; ++tt) {
    int n = n0 + tt * 16 + ln;
    #pragma unroll
    for (int v = 0; v < 8; ++v) {
      int r = row0 + lh * 8 + v;
      P[(size_t)r * NN + n] = f2bf(acc[tt][v] * sv[v]);
    }
  }
}

// ---------------------------------------------------------------------------
// Kernel 4: out = P @ V (B-fragments from Vt rows). f32 output.
// ---------------------------------------------------------------------------
__global__ __launch_bounds__(256) void av_kernel(
    const unsigned short* __restrict__ Pb,
    const unsigned short* __restrict__ Vtb,
    float* __restrict__ out) {
  __shared__ unsigned short ldsA[2 * 128 * LDST];
  __shared__ unsigned short ldsB[2 * 128 * LDST];

  const int lane = threadIdx.x & 31;
  const int w    = threadIdx.x >> 5;
  const int b    = blockIdx.z;
  const unsigned short* P  = Pb  + (size_t)b * NN * NN;
  const unsigned short* Vt = Vtb + (size_t)b * DD * NN;
  float* O = out + (size_t)b * NN * DD;

  v8f zero = {};
  v8f acc[2][4];
  #pragma unroll
  for (int i = 0; i < 2; ++i)
    #pragma unroll
    for (int j = 0; j < 4; ++j) acc[i][j] = zero;

  gemm128(P, Vt, NN, blockIdx.x * 128, blockIdx.y * 128, ldsA, ldsB, acc);

  const int lh = lane >> 4;
  const int ln = lane & 15;
  const int m0 = blockIdx.x * 128 + (w & 3) * 32;
  const int e0 = blockIdx.y * 128 + (w >> 2) * 64;
  #pragma unroll
  for (int i = 0; i < 2; ++i)
    #pragma unroll
    for (int j = 0; j < 4; ++j) {
      int e = e0 + j * 16 + ln;
      #pragma unroll
      for (int v = 0; v < 8; ++v) {
        int m = m0 + i * 16 + lh * 8 + v;
        O[(size_t)m * DD + e] = acc[i][j][v];
      }
    }
}

// ---------------------------------------------------------------------------
extern "C" void kernel_launch(void* const* d_in, const int* in_sizes, int n_in,
                              void* d_out, int out_size, void* d_ws, size_t ws_size,
                              hipStream_t stream) {
  const float* x  = (const float*)d_in[0];
  const float* wq = (const float*)d_in[1];
  const float* wk = (const float*)d_in[2];
  const float* wv = (const float*)d_in[3];
  float* out = (float*)d_out;

  unsigned char* ws = (unsigned char*)d_ws;
  size_t off = 0;
  unsigned short* xb  = (unsigned short*)(ws + off); off += (size_t)BB * NN * DD * 2; // 16 MB
  unsigned short* wqb = (unsigned short*)(ws + off); off += (size_t)DD * DD * 2;      //  2 MB
  unsigned short* wkb = (unsigned short*)(ws + off); off += (size_t)DD * DD * 2;      //  2 MB
  unsigned short* wvb = (unsigned short*)(ws + off); off += (size_t)DD * DD * 2;      //  2 MB
  unsigned short* Qb  = (unsigned short*)(ws + off); off += (size_t)BB * NN * DD * 2; // 16 MB
  unsigned short* Kb  = (unsigned short*)(ws + off); off += (size_t)BB * NN * DD * 2; // 16 MB
  unsigned short* Vtb = (unsigned short*)(ws + off); off += (size_t)BB * DD * NN * 2; // 16 MB
  unsigned short* Pb  = (unsigned short*)(ws + off); off += (size_t)BB * NN * NN * 2; // 32 MB

  // 1) converts
  {
    int n4 = BB * NN * DD / 4;
    cvt_f32_bf16<<<(n4 + 255) / 256, 256, 0, stream>>>(x, xb, n4);
    int m4 = DD * DD / 4;
    cvt_f32_bf16<<<(m4 + 255) / 256, 256, 0, stream>>>(wq, wqb, m4);
    cvt_f32_bf16<<<(m4 + 255) / 256, 256, 0, stream>>>(wk, wkb, m4);
    cvt_f32_bf16<<<(m4 + 255) / 256, 256, 0, stream>>>(wv, wvb, m4);
  }

  // 2) Q, K, V^T projections
  dim3 gp(NN / 128, DD / 128, BB * 3);
  proj_kernel<<<gp, 256, 0, stream>>>(xb, wqb, wkb, wvb, Qb, Kb, Vtb);

  // 3) fused score + softmax -> P (bf16, includes /32)
  dim3 gs(NN / 16, BB, 1);
  score_softmax_kernel<<<gs, 256, 0, stream>>>(Qb, Kb, Pb);

  // 4) out = P @ V
  dim3 ga(NN / 128, DD / 128, BB);
  av_kernel<<<ga, 256, 0, stream>>>(Pb, Vtb, out);
}
